// GraphAttentionLayer_89859305767164
// MI455X (gfx1250) — matrix operand order repaired and checked by reading, not compiled, not run on hardware
//
#include <hip/hip_runtime.h>

// ---------------------------------------------------------------------------
// GraphAttentionLayer for MI455X (gfx1250, wave32, WMMA f32_16x16x32_f16)
//
// Pipeline:
//   1) qkv_gemm  : qkv = x @ in_proj_w^T + b   -> q,k (b,h,n,d) f16 ; v^T (b,h,d,n) f16
//   2) flash_attn: streaming softmax(S^T = K Q^T) ; O^T += V^T P^T  -> attn (b,n,e) f16
//   3) out_gemm  : out = attn @ out_w^T + out_b (fp32)
// Scores are never materialized (would be 512 MB); adj[0] (16 MB) stays in L2.
// ---------------------------------------------------------------------------

#define WPB 8  // waves per 256-thread block

typedef _Float16 v16h __attribute__((ext_vector_type(16)));
typedef _Float16 v8h  __attribute__((ext_vector_type(8)));
typedef float    v8f  __attribute__((ext_vector_type(8)));
typedef float    f4   __attribute__((ext_vector_type(4)));
typedef int      i4   __attribute__((ext_vector_type(4)));

__device__ __forceinline__ float fast_exp2(float x) {
  return __builtin_amdgcn_exp2f(x);   // v_exp_f32
}

__device__ __forceinline__ v8f wmma16(v16h a, v16h b, v8f c) {
  // D = A(16x32 f16) * B(32x16 f16) + C(16x16 f32)
  return __builtin_amdgcn_wmma_f32_16x16x32_f16(false, a, false, b, (short)0, c,
                                                false, false);
}

__device__ __forceinline__ v8h cvt8(f4 a, f4 b) {
  v8h r;
#pragma unroll
  for (int i = 0; i < 4; ++i) { r[i] = (_Float16)a[i]; r[i + 4] = (_Float16)b[i]; }
  return r;
}

__device__ __forceinline__ v16h cat16(v8h a, v8h b) {
  return __builtin_shufflevector(a, b, 0, 1, 2, 3, 4, 5, 6, 7, 8, 9, 10, 11, 12, 13, 14, 15);
}

__device__ __forceinline__ unsigned packh2(float a, float b) {
  union { _Float16 h[2]; unsigned u; } t;
  t.h[0] = (_Float16)a; t.h[1] = (_Float16)b;
  return t.u;
}

// ---------------------------------------------------------------------------
// Kernel 1: QKV projection.  M=16384 tokens, F=384 outputs, K=128.
// One wave = one 16x16 output tile; 4 K-chunks of 32 -> 4 WMMAs.
// ---------------------------------------------------------------------------
__global__ __launch_bounds__(256) void qkv_gemm(const float* __restrict__ x,
                                                const float* __restrict__ w,
                                                const float* __restrict__ bias,
                                                _Float16* __restrict__ qh,
                                                _Float16* __restrict__ kh,
                                                _Float16* __restrict__ vT) {
  const unsigned wid  = blockIdx.x * WPB + (threadIdx.x >> 5);
  const unsigned l15  = threadIdx.x & 15u;
  const bool     hi   = (threadIdx.x & 31u) >= 16u;

  const unsigned mtile = wid & 1023u;   // 1024 token tiles
  const unsigned ftile = wid >> 10;     // 24 output-feature tiles
  const unsigned mbase = mtile << 4;
  const unsigned fbase = ftile << 4;
  const unsigned f     = fbase + l15;

  const float bv = bias[f];
  v8f c = { bv, bv, bv, bv, bv, bv, bv, bv };

  const float* xrow = x + (size_t)(mbase + l15) * 128u;
  const float* wrow = w + (size_t)f * 128u;

#pragma unroll
  for (unsigned e0 = 0; e0 < 128u; e0 += 32u) {
    const unsigned ea = e0 + (hi ? 8u : 0u);    // A lanes lo: K 0-7,16-23 ; hi: 8-15,24-31
    v16h A = cat16(cvt8(*(const f4*)(xrow + ea),      *(const f4*)(xrow + ea + 4)),
                   cvt8(*(const f4*)(xrow + ea + 16), *(const f4*)(xrow + ea + 20)));
    const unsigned eb = e0 + (hi ? 8u : 0u);    // B lanes lo: K 0-7,16-23 ; hi: 8-15,24-31
    v16h Bm = cat16(cvt8(*(const f4*)(wrow + eb),      *(const f4*)(wrow + eb + 4)),
                    cvt8(*(const f4*)(wrow + eb + 16), *(const f4*)(wrow + eb + 20)));
    c = wmma16(A, Bm, c);
  }

  // C layout: VGPR v -> row m=v (+8 for hi lanes), lane -> column f.
  const unsigned sel = f >> 7;           // 0=q 1=k 2=v  (wave-uniform)
  const unsigned hh  = (f & 127u) >> 5;  // head          (wave-uniform)
  const unsigned d   = f & 31u;
  const unsigned m0  = mbase + (hi ? 8u : 0u);
  const unsigned bb  = m0 >> 11;
  const unsigned n0  = m0 & 2047u;
  const unsigned bh4 = bb * 4u + hh;

  if (sel == 0u) {
    _Float16* dst = qh + ((size_t)(bh4 * 2048u + n0) * 32u + d);
#pragma unroll
    for (int v = 0; v < 8; ++v) dst[(size_t)v * 32u] = (_Float16)c[v];
  } else if (sel == 1u) {
    _Float16* dst = kh + ((size_t)(bh4 * 2048u + n0) * 32u + d);
#pragma unroll
    for (int v = 0; v < 8; ++v) dst[(size_t)v * 32u] = (_Float16)c[v];
  } else {
    // v transposed: (b,h,d,n) -> token index is contiguous per lane: one b128 store
    union { _Float16 h[8]; i4 q; } pk;
#pragma unroll
    for (int v = 0; v < 8; ++v) pk.h[v] = (_Float16)c[v];
    *(i4*)(vT + (size_t)(bh4 * 32u + d) * 2048u + n0) = pk.q;
  }
}

// ---------------------------------------------------------------------------
// Kernel 2: flash attention. One wave = one (b,h, 16-query tile).
// S^T = K Q^T keeps query on lanes / key on VGPRs -> in-lane softmax.
// O^T = V^T P^T accumulated in registers; P^T -> B operand via 8 shfl_xor(16).
// ---------------------------------------------------------------------------
__global__ __launch_bounds__(256) void flash_attn(const _Float16* __restrict__ qh,
                                                  const _Float16* __restrict__ kh,
                                                  const _Float16* __restrict__ vT,
                                                  const int* __restrict__ adj0,
                                                  _Float16* __restrict__ attn) {
  const unsigned wid  = blockIdx.x * WPB + (threadIdx.x >> 5);
  const unsigned l15  = threadIdx.x & 15u;
  const bool     hi   = (threadIdx.x & 31u) >= 16u;

  const unsigned qt    = wid & 127u;   // 128 query tiles
  const unsigned bh    = wid >> 7;     // 32 (b,h) pairs
  const unsigned qbase = qt << 4;

  // Q as B operand: lane = query col, K-dim = d (lo lanes d0-7,16-23, hi lanes d8-15,24-31)
  const _Float16* qp = qh + (size_t)(bh * 2048u + qbase + l15) * 32u + (hi ? 8u : 0u);
  const v16h bQ = cat16(*(const v8h*)qp, *(const v8h*)(qp + 16));

  const _Float16* kbp = kh + (size_t)bh * 2048u * 32u;
  const _Float16* vbp = vT + (size_t)bh * 32u * 2048u;
  const int* adjrow = adj0 + (size_t)(qbase + l15) * 2048u;

  const float SC   = 0.17677669529663687f * 1.4426950408889634f; // (1/sqrt(D))*log2(e)
  const float NEG2 = -1.442695e9f;                               // -1e9*log2(e)

  float m = -3.0e38f;
  float l = 0.0f;
  v8f o0 = {0, 0, 0, 0, 0, 0, 0, 0};   // O^T rows d=0..15
  v8f o1 = {0, 0, 0, 0, 0, 0, 0, 0};   // O^T rows d=16..31

  for (unsigned kb = 0; kb < 2048u; kb += 32u) {
    // ---- S^T tiles: (16 keys x 16 queries) each, K-dim = D = 32 -> 1 WMMA each
    const _Float16* kp0 = kbp + (size_t)(kb + l15) * 32u + (hi ? 8u : 0u);
    const _Float16* kp1 = kp0 + 16u * 32u;
    v8f z = {0, 0, 0, 0, 0, 0, 0, 0};
    v8f s0 = wmma16(cat16(*(const v8h*)kp0, *(const v8h*)(kp0 + 16)), bQ, z);
    v8f s1 = wmma16(cat16(*(const v8h*)kp1, *(const v8h*)(kp1 + 16)), bQ, z);

    // ---- graph mask + scale (log2 domain). adj ints contiguous per lane.
    const i4 a00 = *(const i4*)(adjrow + kb + (hi ? 8u : 0u));
    const i4 a01 = *(const i4*)(adjrow + kb + (hi ? 8u : 0u) + 4);
    const i4 a10 = *(const i4*)(adjrow + kb + 16u + (hi ? 8u : 0u));
    const i4 a11 = *(const i4*)(adjrow + kb + 16u + (hi ? 8u : 0u) + 4);

    float lg0[8], lg1[8];
#pragma unroll
    for (int j = 0; j < 4; ++j) {
      lg0[j]     = s0[j]     * SC + (a00[j] ? 0.0f : NEG2);
      lg0[j + 4] = s0[j + 4] * SC + (a01[j] ? 0.0f : NEG2);
      lg1[j]     = s1[j]     * SC + (a10[j] ? 0.0f : NEG2);
      lg1[j + 4] = s1[j + 4] * SC + (a11[j] ? 0.0f : NEG2);
    }

    // ---- online softmax stats: reduce over keys = over registers, then one
    //      shfl_xor(16) to merge the two half-wave key groups of each query.
    float cm = lg0[0];
#pragma unroll
    for (int j = 1; j < 8; ++j) cm = fmaxf(cm, lg0[j]);
#pragma unroll
    for (int j = 0; j < 8; ++j) cm = fmaxf(cm, lg1[j]);
    cm = fmaxf(cm, __shfl_xor(cm, 16));
    const float mnew  = fmaxf(m, cm);
    const float alpha = fast_exp2(m - mnew);

    float p0[8], p1[8];
    float rs = 0.0f;
#pragma unroll
    for (int j = 0; j < 8; ++j) { p0[j] = fast_exp2(lg0[j] - mnew); rs += p0[j]; }
#pragma unroll
    for (int j = 0; j < 8; ++j) { p1[j] = fast_exp2(lg1[j] - mnew); rs += p1[j]; }
    rs += __shfl_xor(rs, 16);
    l  = l * alpha + rs;
    m  = mnew;
    o0 = o0 * alpha;
    o1 = o1 * alpha;

    // ---- P^T (32 keys x 16 queries) as WMMA B operand.
    // B lanes lo need keys 0-15 (own t0 + partner's t0); hi need 16-31.
    unsigned t0d[4], t1d[4];
#pragma unroll
    for (int j = 0; j < 4; ++j) {
      t0d[j] = packh2(p0[2 * j], p0[2 * j + 1]);
      t1d[j] = packh2(p1[2 * j], p1[2 * j + 1]);
    }
    union { unsigned u[8]; v16h v; } bp;
#pragma unroll
    for (int j = 0; j < 4; ++j) {
      const unsigned x0 = (unsigned)__shfl_xor((int)t0d[j], 16);
      const unsigned x1 = (unsigned)__shfl_xor((int)t1d[j], 16);
      bp.u[j]     = hi ? x1 : t0d[j];
      bp.u[j + 4] = hi ? t1d[j] : x0;
    }
    const v16h bP = bp.v;

    // ---- O^T += V^T * P^T   (A = V^T rows d, K-dim = 32 keys)
    const _Float16* vp0 = vbp + (size_t)l15 * 2048u + kb + (hi ? 16u : 0u);
    const _Float16* vp1 = vp0 + 16u * 2048u;
    o0 = wmma16(cat16(*(const v8h*)vp0, *(const v8h*)(vp0 + 8)), bP, o0);
    o1 = wmma16(cat16(*(const v8h*)vp1, *(const v8h*)(vp1 + 8)), bP, o1);
  }

  // ---- normalize + write (b,n,e) f16; per lane: 8 consecutive halves -> b128
  const float inv = 1.0f / l;
  const unsigned bb = bh >> 2, hh = bh & 3u;
  _Float16* orow = attn + (size_t)(bb * 2048u + qbase + l15) * 128u + hh * 32u + (hi ? 8u : 0u);
  union { _Float16 h[8]; i4 q; } pk0, pk1;
#pragma unroll
  for (int v = 0; v < 8; ++v) {
    pk0.h[v] = (_Float16)(o0[v] * inv);
    pk1.h[v] = (_Float16)(o1[v] * inv);
  }
  *(i4*)orow        = pk0.q;
  *(i4*)(orow + 16) = pk1.q;
}

// ---------------------------------------------------------------------------
// Kernel 3: output projection.  M=16384, F=128, K=128. fp32 result to d_out.
// ---------------------------------------------------------------------------
__global__ __launch_bounds__(256) void out_gemm(const _Float16* __restrict__ attn,
                                                const float* __restrict__ w,
                                                const float* __restrict__ bias,
                                                float* __restrict__ out) {
  const unsigned wid  = blockIdx.x * WPB + (threadIdx.x >> 5);
  const unsigned l15  = threadIdx.x & 15u;
  const bool     hi   = (threadIdx.x & 31u) >= 16u;

  const unsigned mtile = wid & 1023u;
  const unsigned ftile = wid >> 10;     // 0..7
  const unsigned mbase = mtile << 4;
  const unsigned f     = (ftile << 4) + l15;

  const float bv = bias[f];
  v8f c = { bv, bv, bv, bv, bv, bv, bv, bv };

  const _Float16* arow = attn + (size_t)(mbase + l15) * 128u;
  const float*    wrow = w + (size_t)f * 128u;

#pragma unroll
  for (unsigned e0 = 0; e0 < 128u; e0 += 32u) {
    const unsigned ea = e0 + (hi ? 8u : 0u);
    v16h A = cat16(*(const v8h*)(arow + ea), *(const v8h*)(arow + ea + 16));
    const unsigned eb = e0 + (hi ? 8u : 0u);
    v16h Bm = cat16(cvt8(*(const f4*)(wrow + eb),      *(const f4*)(wrow + eb + 4)),
                    cvt8(*(const f4*)(wrow + eb + 16), *(const f4*)(wrow + eb + 20)));
    c = wmma16(A, Bm, c);
  }

  float* orow = out + (size_t)(mbase + (hi ? 8u : 0u)) * 128u + f;
#pragma unroll
  for (int v = 0; v < 8; ++v) orow[(size_t)v * 128u] = c[v];
}

// ---------------------------------------------------------------------------
extern "C" void kernel_launch(void* const* d_in, const int* in_sizes, int n_in,
                              void* d_out, int out_size, void* d_ws, size_t ws_size,
                              hipStream_t stream) {
  const float* x    = (const float*)d_in[0];   // (8,2048,128)
  const int*   adj  = (const int*)d_in[1];     // (8,2048,2048) -> only adj[0] used
  const float* ipw  = (const float*)d_in[2];   // (384,128)
  const float* ipb  = (const float*)d_in[3];   // (384,)
  const float* ow   = (const float*)d_in[4];   // (128,128)
  const float* ob   = (const float*)d_in[5];   // (128,)
  float* out = (float*)d_out;

  char* ws = (char*)d_ws;                       // 16 MB used
  _Float16* qh   = (_Float16*)(ws);             // (b,h,n,d) f16, 4 MB
  _Float16* kh   = (_Float16*)(ws + (4u  << 20));
  _Float16* vT   = (_Float16*)(ws + (8u  << 20)); // (b,h,d,n) f16
  _Float16* attn = (_Float16*)(ws + (12u << 20)); // (b,n,e) f16

  // 1) QKV: 1024 m-tiles x 24 f-tiles = 24576 waves
  qkv_gemm<<<24576 / WPB, 256, 0, stream>>>(x, ipw, ipb, qh, kh, vT);
  // 2) attention: 32 (b,h) x 128 q-tiles = 4096 waves
  flash_attn<<<4096 / WPB, 256, 0, stream>>>(qh, kh, vT, adj, attn);
  // 3) out projection: 1024 m-tiles x 8 f-tiles = 8192 waves
  out_gemm<<<8192 / WPB, 256, 0, stream>>>(attn, ow, ob, out);
}
